// SelfAttention_52321291600035
// MI455X (gfx1250) — compile-verified, matrix operand order checked by reading, not run on hardware
//
#include <hip/hip_runtime.h>
#include <hip/hip_bf16.h>

#define B_  4
#define S_  2048
#define D_  1024
#define H_  16
#define HD_ 64
#define P_  256
#define R_  128

typedef __attribute__((ext_vector_type(16))) _Float16 v16h;
typedef __attribute__((ext_vector_type(8)))  _Float16 v8h;
typedef __attribute__((ext_vector_type(4)))  _Float16 v4h;
typedef __attribute__((ext_vector_type(8)))  float    v8f;
typedef __attribute__((ext_vector_type(4)))  int      v4i_t;

union F8  { v8f  v; float    f[8];  };
union H16 { v16h v; _Float16 h[16]; };
union H8  { v8h  v; _Float16 h[8];  };

#if defined(__has_builtin)
#if __has_builtin(__builtin_amdgcn_global_load_async_to_lds_b128)
#define HAVE_ASYNC_LDS 1
// builtin expects (AS1 v4i*, AS3 v4i*, imm offset, imm cpol) per compiler diagnostic
#define ASYNC_GPTR(p) ((__attribute__((address_space(1))) v4i_t*)(p))
#define ASYNC_LPTR(p) ((__attribute__((address_space(3))) v4i_t*)(p))
#endif
#endif

// Load one 16x32 f16 A-pattern fragment from row-major memory (global or LDS).
// Lane L: row = L&15, kb = 8*(L>>4); elements 0..7 = K kb..kb+7, 8..15 = K 16+kb..16+kb+7.
__device__ __forceinline__ v16h frag_ld(const _Float16* base, int ld) {
    const int lane = threadIdx.x & 31;
    const _Float16* r = base + (size_t)(lane & 15) * ld + ((lane >> 4) << 3);
    v8h lo = *(const v8h*)(r);
    v8h hi = *(const v8h*)(r + 16);
    return __builtin_shufflevector(lo, hi, 0,1,2,3,4,5,6,7,8,9,10,11,12,13,14,15);
}

__device__ __forceinline__ v8f wmma32(v16h a, v16h b, v8f c) {
    return __builtin_amdgcn_wmma_f32_16x16x32_f16(false, a, false, b, (short)0, c, false, false);
}

// Store a 16x16 f32 C/D fragment: lane = col (mod 16), VGPR v = row v (+8 for lanes>=16).
template<int HALF>
__device__ __forceinline__ void frag_st(void* Cv, int ldc, int m, int n, v8f c) {
    const int lane = threadIdx.x & 31;
    const int col  = n + (lane & 15);
    const int rb   = m + ((lane >> 4) << 3);
    F8 u; u.v = c;
    if (HALF) {
        _Float16* C = (_Float16*)Cv;
#pragma unroll
        for (int v = 0; v < 8; ++v) C[(size_t)(rb + v) * ldc + col] = (_Float16)u.f[v];
    } else {
        float* C = (float*)Cv;
#pragma unroll
        for (int v = 0; v < 8; ++v) C[(size_t)(rb + v) * ldc + col] = u.f[v];
    }
}

// ---------------------------------------------------------------------------
// fp32 -> fp16 conversion, 4 elements/thread (all sizes are multiples of 4)
// ---------------------------------------------------------------------------
__global__ __launch_bounds__(256)
void cvt_f32_f16(const float* __restrict__ in, _Float16* __restrict__ out, int n4) {
    int i = blockIdx.x * blockDim.x + threadIdx.x;
    if (i < n4) {
        const float4 x = ((const float4*)in)[i];
        v4h y;
        y[0] = (_Float16)x.x; y[1] = (_Float16)x.y;
        y[2] = (_Float16)x.z; y[3] = (_Float16)x.w;
        ((v4h*)out)[i] = y;
    }
}

// ---------------------------------------------------------------------------
// C[M,N] = A[M,K] * Bm[N,K]^T   (both K-contiguous; dims divide block tile)
// Block = 8 waves (4M x 2N); block tile 128x128; wave tile 32(M) x 64(N):
// 8 WMMAs per 6 fragment loads per k-step.
// blockIdx.z offsets (elements): A += z*azoff, C += z*czoff  (per-head rel GEMM)
// ---------------------------------------------------------------------------
template<int HALF_OUT>
__global__ __launch_bounds__(256)
void gemm_tn(const _Float16* __restrict__ A, int lda, int azoff,
             const _Float16* __restrict__ Bm, int ldb,
             void* __restrict__ Cv, int ldc, int czoff, int K) {
    A += (size_t)blockIdx.z * azoff;
    const int wave = threadIdx.x >> 5;
    const int m0 = blockIdx.y * 128 + (wave >> 1) * 32;
    const int n0 = blockIdx.x * 128 + (wave & 1) * 64;
    v8f acc[2][4];
#pragma unroll
    for (int i = 0; i < 2; ++i)
#pragma unroll
        for (int j = 0; j < 4; ++j) acc[i][j] = (v8f){};
    for (int k = 0; k < K; k += 32) {
        v16h a0 = frag_ld(A + (size_t)m0 * lda + k,        lda);
        v16h a1 = frag_ld(A + (size_t)(m0 + 16) * lda + k, lda);
#pragma unroll
        for (int j = 0; j < 4; ++j) {
            v16h b = frag_ld(Bm + (size_t)(n0 + j * 16) * ldb + k, ldb);
            acc[0][j] = wmma32(a0, b, acc[0][j]);
            acc[1][j] = wmma32(a1, b, acc[1][j]);
        }
    }
    char* Cz = (char*)Cv + (size_t)blockIdx.z * czoff * (HALF_OUT ? 2 : 4);
#pragma unroll
    for (int i = 0; i < 2; ++i)
#pragma unroll
        for (int j = 0; j < 4; ++j)
            frag_st<HALF_OUT>(Cz, ldc, m0 + i * 16, n0 + j * 16, acc[i][j]);
}

// ---------------------------------------------------------------------------
// Flash attention, transposed-score formulation:
//   S^T = K_tile @ Q^T  -> C-layout: lane = query, VGPR = key
//   O^T = V^T  @ P^T    -> P^T B-fragment == per-lane registers (no LDS repack)
// Bias added twice (reference double-adds att_pos); far tiles use the clamped
// constant rel value (one load per lane) instead of per-element gathers.
// grid = (S/128, B*H); block = 256 (8 waves, 16 query cols per wave).
// ---------------------------------------------------------------------------
__global__ __launch_bounds__(256)
void attn_kernel(const _Float16* __restrict__ qh, const _Float16* __restrict__ kh,
                 const _Float16* __restrict__ vh, const _Float16* __restrict__ relh,
                 _Float16* __restrict__ yh) {
    __shared__ __align__(16) _Float16 Kt[64][72];   // [key][d]
    __shared__ __align__(16) _Float16 Vt[64][72];   // [d][key] (transposed)

    const int bh    = blockIdx.y;
    const int b     = bh >> 4;
    const int h     = bh & 15;
    const int wave  = threadIdx.x >> 5;
    const int lane  = threadIdx.x & 31;
    const int qcol  = lane & 15;                    // this lane's query (mod 16)
    const int rhalf = (lane >> 4) << 3;             // 0 or 8 (C-layout row half / kb)
    const int qrow0 = blockIdx.x * 128 + wave * 16;
    const int qs    = qrow0 + qcol;                 // this lane's query row

    // Q fragments (B operand: A-pattern of Q)
    const _Float16* qp = qh + ((size_t)(b * S_ + qrow0)) * D_ + h * HD_;
    v16h qb0 = frag_ld(qp,      D_);
    v16h qb1 = frag_ld(qp + 32, D_);

    // per-lane rel row + clamped-end biases
    const _Float16* relrow = relh + ((size_t)(b * S_ + qs) * H_ + h) * P_;
    const float blo = 2.0f * (float)relrow[R_];       // d clipped to -128 -> idx 128
    const float bhi = 2.0f * (float)relrow[R_ - 1];   // d clipped to +127 -> idx 127

    F8 o[4];                                          // O^T: rows = out channel, cols = query
#pragma unroll
    for (int f = 0; f < 4; ++f)
#pragma unroll
        for (int v = 0; v < 8; ++v) o[f].f[v] = 0.f;
    float mrun = -3.0e38f, lrun = 0.f;
    const float scale = 0.125f;                       // 1/sqrt(64)

    // cooperative tile-load mapping: 256 threads * 16 halves = 64x64 tile
    const int li  = threadIdx.x * 16;
    const int tlr = li >> 6;                          // 0..63
    const int tlc = li & 63;                          // 0,16,32,48

    for (int t0 = 0; t0 < S_; t0 += 64) {
        __syncthreads();
        {
            const _Float16* ks = kh + ((size_t)(b * S_ + t0 + tlr)) * D_ + h * HD_ + tlc;
#if defined(HAVE_ASYNC_LDS)
            __builtin_amdgcn_global_load_async_to_lds_b128(
                ASYNC_GPTR(ks),     ASYNC_LPTR(&Kt[tlr][tlc]),     0, 0);
            __builtin_amdgcn_global_load_async_to_lds_b128(
                ASYNC_GPTR(ks + 8), ASYNC_LPTR(&Kt[tlr][tlc + 8]), 0, 0);
#else
            *(v8h*)&Kt[tlr][tlc]     = *(const v8h*)ks;
            *(v8h*)&Kt[tlr][tlc + 8] = *(const v8h*)(ks + 8);
#endif
            const _Float16* vs = vh + ((size_t)(b * S_ + t0 + tlr)) * D_ + h * HD_ + tlc;
            v8h x0 = *(const v8h*)vs;
            v8h x1 = *(const v8h*)(vs + 8);
#pragma unroll
            for (int j = 0; j < 8; ++j) Vt[tlc + j][tlr]     = x0[j];
#pragma unroll
            for (int j = 0; j < 8; ++j) Vt[tlc + 8 + j][tlr] = x1[j];
            if (t0 + 64 < S_) {
                __builtin_prefetch(ks + 64 * D_, 0, 0);
                __builtin_prefetch(vs + 64 * D_, 0, 0);
            }
        }
#if defined(HAVE_ASYNC_LDS)
        asm volatile("s_wait_asynccnt 0" ::: "memory");
#endif
        __syncthreads();

        // S^T: 64 keys x 16 queries (4 fragments over keys, K=64 -> 2 WMMAs each)
        F8 sc[4];
#pragma unroll
        for (int f = 0; f < 4; ++f) {
            v16h ka0 = frag_ld(&Kt[f * 16][0],  72);
            v16h ka1 = frag_ld(&Kt[f * 16][32], 72);
            v8f s = (v8f){};
            s = wmma32(ka0, qb0, s);
            s = wmma32(ka1, qb1, s);
            sc[f].v = s;
        }

        // bias (2*rel, clamped fast paths) + scale; element (f,v): key = t0+f*16+rhalf+v
        if (t0 + 192 <= qrow0) {                      // every d <= -129
#pragma unroll
            for (int f = 0; f < 4; ++f)
#pragma unroll
                for (int v = 0; v < 8; ++v) sc[f].f[v] = (sc[f].f[v] + blo) * scale;
        } else if (t0 >= qrow0 + 15 + R_) {           // every d >= 128
#pragma unroll
            for (int f = 0; f < 4; ++f)
#pragma unroll
                for (int v = 0; v < 8; ++v) sc[f].f[v] = (sc[f].f[v] + bhi) * scale;
        } else {
#pragma unroll
            for (int f = 0; f < 4; ++f)
#pragma unroll
                for (int v = 0; v < 8; ++v) {
                    int d = (t0 + f * 16 + rhalf + v) - qs;
                    d = d < -R_ ? -R_ : (d > R_ - 1 ? R_ - 1 : d);
                    const float bias = 2.0f * (float)relrow[d & (P_ - 1)];
                    sc[f].f[v] = (sc[f].f[v] + bias) * scale;
                }
        }

        // online softmax over keys: all of a query's scores live in lanes L and L^16
        float mv = sc[0].f[0];
#pragma unroll
        for (int f = 0; f < 4; ++f)
#pragma unroll
            for (int v = 0; v < 8; ++v) mv = fmaxf(mv, sc[f].f[v]);
        mv = fmaxf(mv, __shfl_xor(mv, 16, 32));
        const float mt = fmaxf(mv, mrun);

        float rs = 0.f;
#pragma unroll
        for (int f = 0; f < 4; ++f)
#pragma unroll
            for (int v = 0; v < 8; ++v) {
                const float p = __expf(sc[f].f[v] - mt);
                sc[f].f[v] = p;
                rs += p;
            }
        rs += __shfl_xor(rs, 16, 32);

        const float alpha = __expf(mrun - mt);
        lrun = lrun * alpha + rs;
        mrun = mt;
#pragma unroll
        for (int f = 0; f < 4; ++f)
#pragma unroll
            for (int v = 0; v < 8; ++v) o[f].f[v] *= alpha;

        // P^T B-fragments straight from registers: chunk c covers keys 32c..32c+31;
        // lane holds exactly the kb-half it must supply (rhalf == kb).
        H16 pb[2];
#pragma unroll
        for (int c = 0; c < 2; ++c)
#pragma unroll
            for (int j = 0; j < 8; ++j) {
                pb[c].h[j]     = (_Float16)sc[2 * c].f[j];
                pb[c].h[8 + j] = (_Float16)sc[2 * c + 1].f[j];
            }

        // O^T += V^T @ P^T : A = V^T rows (out channels), K = keys
#pragma unroll
        for (int f = 0; f < 4; ++f) {
            v16h va0 = frag_ld(&Vt[f * 16][0],  72);
            v16h va1 = frag_ld(&Vt[f * 16][32], 72);
            o[f].v = wmma32(va0, pb[0].v, o[f].v);
            o[f].v = wmma32(va1, pb[1].v, o[f].v);
        }
    }

    // finalize: per-lane 1/l, store O^T (rows = channels, col = this lane's query)
    const float inv = 1.0f / lrun;
    _Float16* yp = yh + ((size_t)(b * S_ + qs)) * D_ + h * HD_;
#pragma unroll
    for (int f = 0; f < 4; ++f) {
        H8 pk;
#pragma unroll
        for (int v = 0; v < 8; ++v) pk.h[v] = (_Float16)(o[f].f[v] * inv);
        *(v8h*)(yp + f * 16 + rhalf) = pk.v;
    }
}

// ---------------------------------------------------------------------------

extern "C" void kernel_launch(void* const* d_in, const int* in_sizes, int n_in,
                              void* d_out, int out_size, void* d_ws, size_t ws_size,
                              hipStream_t stream) {
    (void)in_sizes; (void)n_in; (void)out_size; (void)ws_size;
    const float* x  = (const float*)d_in[0];
    const float* Wq = (const float*)d_in[1];
    const float* Wk = (const float*)d_in[2];
    const float* Wv = (const float*)d_in[3];
    const float* Wp = (const float*)d_in[4];
    const float* pe = (const float*)d_in[5];

    char* w = (char*)d_ws;
    auto alloc = [&](size_t bytes) -> void* {
        void* p = (void*)w;
        w += (bytes + 255) & ~(size_t)255;
        return p;
    };
    const size_t BS = (size_t)B_ * S_;   // 8192
    _Float16* xh   = (_Float16*)alloc(BS * D_ * 2);
    _Float16* wqh  = (_Float16*)alloc((size_t)D_ * D_ * 2);
    _Float16* wkh  = (_Float16*)alloc((size_t)D_ * D_ * 2);
    _Float16* wvh  = (_Float16*)alloc((size_t)D_ * D_ * 2);
    _Float16* wph  = (_Float16*)alloc((size_t)D_ * D_ * 2);
    _Float16* peh  = (_Float16*)alloc((size_t)P_ * HD_ * 2);
    _Float16* q16  = (_Float16*)alloc(BS * D_ * 2);
    _Float16* k16  = (_Float16*)alloc(BS * D_ * 2);
    _Float16* v16  = (_Float16*)alloc(BS * D_ * 2);
    _Float16* relh = (_Float16*)alloc(BS * (size_t)H_ * P_ * 2);  // [B,S,H,P]
    _Float16* y16  = (_Float16*)alloc(BS * D_ * 2);

    const dim3 blk(256);
    { int n4 = (int)(BS * D_ / 4);
      cvt_f32_f16<<<dim3((n4 + 255) / 256), blk, 0, stream>>>(x, xh, n4); }
    { int n4 = D_ * D_ / 4;
      dim3 g((n4 + 255) / 256);
      cvt_f32_f16<<<g, blk, 0, stream>>>(Wq, wqh, n4);
      cvt_f32_f16<<<g, blk, 0, stream>>>(Wk, wkh, n4);
      cvt_f32_f16<<<g, blk, 0, stream>>>(Wv, wvh, n4);
      cvt_f32_f16<<<g, blk, 0, stream>>>(Wp, wph, n4); }
    { int n4 = P_ * HD_ / 4;
      cvt_f32_f16<<<dim3((n4 + 255) / 256), blk, 0, stream>>>(pe, peh, n4); }

    // q/k/v = x @ W^T   (M=8192, N=1024, K=1024)
    const dim3 gqkv(D_ / 128, BS / 128, 1);
    gemm_tn<1><<<gqkv, blk, 0, stream>>>(xh, D_, 0, wqh, D_, q16, D_, 0, D_);
    gemm_tn<1><<<gqkv, blk, 0, stream>>>(xh, D_, 0, wkh, D_, k16, D_, 0, D_);
    gemm_tn<1><<<gqkv, blk, 0, stream>>>(xh, D_, 0, wvh, D_, v16, D_, 0, D_);

    // rel[b,s,h,:] = q_head @ pos^T   (per head via blockIdx.z; M=8192, N=256, K=64)
    const dim3 grel(P_ / 128, BS / 128, H_);
    gemm_tn<1><<<grel, blk, 0, stream>>>(q16, D_, HD_, peh, HD_, relh, H_ * P_, P_, HD_);

    // flash attention
    const dim3 gat(S_ / 128, B_ * H_, 1);
    attn_kernel<<<gat, blk, 0, stream>>>(q16, k16, v16, relh, y16);

    // out = y @ Wp^T  (f32 output)
    gemm_tn<0><<<gqkv, blk, 0, stream>>>(y16, D_, 0, wph, D_, d_out, D_, 0, D_);
}
